// CGCEncoder_74028056314066
// MI455X (gfx1250) — matrix operand order repaired and checked by reading, not compiled, not run on hardware
//
#include <hip/hip_runtime.h>
#include <math.h>

typedef _Float16 half_t;
typedef __attribute__((ext_vector_type(16))) _Float16 v16h;
typedef __attribute__((ext_vector_type(8)))  _Float16 v8h;
typedef __attribute__((ext_vector_type(8)))  float    v8f;

#define N_NODES 50000
#define N_EDGES 800000
#define D_FEAT  64
#define D_EDGE  32
#define Z_DIM   160   // 2*64 + 32

// ---------------------------------------------------------------------------
// Gather one contiguous 8-element K-run of z = [x[dst] | x[src] | edge_attr]
// as f16. Runs are 8-aligned so they never straddle the 64/128 boundaries.
// ---------------------------------------------------------------------------
__device__ inline v8h load_z_run8(const half_t* __restrict__ h16,
                                  const float* __restrict__ ea,
                                  int dstI, int srcI, int e, int K) {
  if (K < 64)  return *(const v8h*)(h16 + (size_t)dstI * D_FEAT + K);
  if (K < 128) return *(const v8h*)(h16 + (size_t)srcI * D_FEAT + (K - 64));
  const float* p = ea + (size_t)e * D_EDGE + (K - 128);
  float4 f0 = *(const float4*)(p);
  float4 f1 = *(const float4*)(p + 4);
  v8h r;
  r[0] = (half_t)f0.x; r[1] = (half_t)f0.y; r[2] = (half_t)f0.z; r[3] = (half_t)f0.w;
  r[4] = (half_t)f1.x; r[5] = (half_t)f1.y; r[6] = (half_t)f1.z; r[7] = (half_t)f1.w;
  return r;
}

// ---------------------------------------------------------------------------
// Edge kernel: per 16-edge tile, msg = sigmoid(z@Wf+bf)*softplus(z@Ws+bs) via
// v_wmma_f32_16x16x32_f16; scatter-add into agg with f32 atomics.
// Weights staged once per block into LDS, pre-swizzled into B-fragment order
// (lane<16: K k0+0..7,16..23, lane>=16: K k0+8..15,24..31, col = nt*16 + lane%16) so each
// B operand is one contiguous 32B LDS read. The fragment index is laundered
// through an empty asm per tile so LICM cannot hoist the 40 B fragments into
// (or spill them out of) registers: B stays in LDS, waves stay small (~128
// VGPRs), occupancy stays high to hide L2 gather + atomic latency.
// ---------------------------------------------------------------------------
__global__ __launch_bounds__(256, 4) void cgconv_edge_kernel(
    const half_t* __restrict__ h16, const int* __restrict__ ei,
    const float* __restrict__ ea,
    const float* __restrict__ Wf, const float* __restrict__ bf,
    const float* __restrict__ Ws, const float* __restrict__ bs,
    float* __restrict__ agg)
{
  __shared__ v16h wlds[2 * 5 * 4 * 32];   // [mat][kstep][ntile][lane] = 40 KB
  __shared__ float bfs[D_FEAT], bss[D_FEAT];

  const int tid = threadIdx.x;

  // Cooperative weight stage: 2 matrices x 160 K-rows, swizzled to fragments.
  for (int idx = tid; idx < 2 * Z_DIM; idx += 256) {
    const int mat = idx / Z_DIM;
    const int K   = idx % Z_DIM;
    const float* Wsrc = mat ? Ws : Wf;
    const int kstep = K >> 5;
    const int r     = K & 31;
    const int rh    = ((r >> 3) & 1) * 16;
    const int j     = (r & 7) | ((r >> 4) << 3);
    for (int c = 0; c < D_FEAT; ++c) {
      wlds[((mat * 5 + kstep) * 4 + (c >> 4)) * 32 + rh + (c & 15)][j] =
          (half_t)Wsrc[K * D_FEAT + c];
    }
  }
  if (tid < D_FEAT) { bfs[tid] = bf[tid]; bss[tid] = bs[tid]; }
  __syncthreads();

  const int lane = tid & 31;
  const int l16  = lane & 15;
  const int hi   = lane >> 4;
  const int gwave  = blockIdx.x * 8 + (tid >> 5);
  const int nwaves = gridDim.x * 8;
  const int ntiles = N_EDGES / 16;

  for (int t = gwave; t < ntiles; t += nwaves) {
    const int e    = t * 16 + l16;     // this lane's A-row edge
    const int srcI = ei[e];            // edge_index[0] = source
    const int dstI = ei[N_EDGES + e];  // edge_index[1] = target

    // Opaque copy of the lane index: defeats LICM so B fragments are
    // re-read from LDS each tile instead of being hoisted/spilled.
    int lane_dyn = lane;
    asm volatile("" : "+v"(lane_dyn));

    v8f facc[4] = {};
    v8f sacc[4] = {};

#pragma unroll
    for (int ks = 0; ks < 5; ++ks) {   // K = 160 in 5 steps of 32
      const int k0 = ks * 32;
      // A fragment per ISA 16-bit 16x32 layout.
      v8h r0 = load_z_run8(h16, ea, dstI, srcI, e, k0 + 8 * hi);
      v8h r1 = load_z_run8(h16, ea, dstI, srcI, e, k0 + 16 + 8 * hi);
      v16h a;
#pragma unroll
      for (int j = 0; j < 8; ++j) { a[j] = r0[j]; a[8 + j] = r1[j]; }

#pragma unroll
      for (int nt = 0; nt < 4; ++nt) {
        v16h bF = wlds[((0 * 5 + ks) * 4 + nt) * 32 + lane_dyn];
        v16h bS = wlds[((1 * 5 + ks) * 4 + nt) * 32 + lane_dyn];
        facc[nt] = __builtin_amdgcn_wmma_f32_16x16x32_f16(
            false, a, false, bF, (short)0, facc[nt], false, false);
        sacc[nt] = __builtin_amdgcn_wmma_f32_16x16x32_f16(
            false, a, false, bS, (short)0, sacc[nt], false, false);
      }
    }

    // C/D layout: VGPR i -> row i (lanes 0-15) / row i+8 (lanes 16-31),
    // col = lane%16 in each 16-wide N tile. dst index broadcast via shuffle.
#pragma unroll
    for (int i = 0; i < 8; ++i) {
      const int dm = __shfl(dstI, i + 8 * hi, 32);
#pragma unroll
      for (int nt = 0; nt < 4; ++nt) {
        const int col = nt * 16 + l16;
        const float f = facc[nt][i] + bfs[col];
        const float s = sacc[nt][i] + bss[col];
        const float sig = 1.0f / (1.0f + __expf(-f));
        const float sp  = (s > 20.0f) ? s : log1pf(__expf(s));
        atomicAdd(&agg[(size_t)dm * D_FEAT + col], sig * sp);
      }
    }
  }
}

// fp32 -> f16 feature mirror (layer-0 input prep)
__global__ void convert_f32_to_f16(const float* __restrict__ in,
                                   half_t* __restrict__ out, int n) {
  int i = blockIdx.x * blockDim.x + threadIdx.x;
  if (i < n) out[i] = (half_t)in[i];
}

// h = relu(prev + agg); refresh both fp32 state and f16 gather mirror
__global__ void node_relu_kernel(const float* __restrict__ prev,
                                 const float* __restrict__ agg,
                                 float* __restrict__ out32,
                                 half_t* __restrict__ out16, int n) {
  int i = blockIdx.x * blockDim.x + threadIdx.x;
  if (i < n) {
    float v = prev[i] + agg[i];
    v = v > 0.0f ? v : 0.0f;
    out32[i] = v;
    out16[i] = (half_t)v;
  }
}

// Final: softmax(prev + agg) over 64 features; one wave32 per node (2 feat/lane)
__global__ void node_softmax_kernel(const float* __restrict__ prev,
                                    const float* __restrict__ agg,
                                    float* __restrict__ out, int n_nodes) {
  int wave = (blockIdx.x * blockDim.x + threadIdx.x) >> 5;
  int lane = threadIdx.x & 31;
  if (wave >= n_nodes) return;
  size_t base = (size_t)wave * D_FEAT;
  float v0 = prev[base + lane]      + agg[base + lane];
  float v1 = prev[base + lane + 32] + agg[base + lane + 32];
  float mx = fmaxf(v0, v1);
#pragma unroll
  for (int off = 16; off > 0; off >>= 1)
    mx = fmaxf(mx, __shfl_xor(mx, off, 32));
  float e0 = __expf(v0 - mx), e1 = __expf(v1 - mx);
  float sum = e0 + e1;
#pragma unroll
  for (int off = 16; off > 0; off >>= 1)
    sum += __shfl_xor(sum, off, 32);
  float inv = 1.0f / sum;
  out[base + lane]      = e0 * inv;
  out[base + lane + 32] = e1 * inv;
}

extern "C" void kernel_launch(void* const* d_in, const int* in_sizes, int n_in,
                              void* d_out, int out_size, void* d_ws, size_t ws_size,
                              hipStream_t stream) {
  (void)in_sizes; (void)n_in; (void)out_size; (void)ws_size;

  const float* x  = (const float*)d_in[0];
  const int*   ei = (const int*)d_in[1];      // [2, E] int32: row0=src, row1=dst
  const float* ea = (const float*)d_in[2];

  struct WSet { const float *Wf, *bf, *Ws, *bs; };
  WSet L1 = {(const float*)d_in[3],  (const float*)d_in[4],
             (const float*)d_in[5],  (const float*)d_in[6]};
  WSet L2 = {(const float*)d_in[7],  (const float*)d_in[8],
             (const float*)d_in[9],  (const float*)d_in[10]};
  WSet L3 = {(const float*)d_in[11], (const float*)d_in[12],
             (const float*)d_in[13], (const float*)d_in[14]};
  WSet apps[4] = {L1, L2, L2, L3};   // conv2 applied twice with shared weights

  const int nfeat = N_NODES * D_FEAT;
  half_t* h16 = (half_t*)d_ws;                                 // 6.4 MB
  float*  agg = (float*)((char*)d_ws + (size_t)nfeat * sizeof(half_t)); // 12.8 MB
  float*  out = (float*)d_out;                                 // running fp32 h

  convert_f32_to_f16<<<(nfeat + 255) / 256, 256, 0, stream>>>(x, h16, nfeat);

  for (int app = 0; app < 4; ++app) {
    hipMemsetAsync(agg, 0, (size_t)nfeat * sizeof(float), stream);
    cgconv_edge_kernel<<<1250, 256, 0, stream>>>(
        h16, ei, ea, apps[app].Wf, apps[app].bf, apps[app].Ws, apps[app].bs, agg);
    const float* prev = (app == 0) ? x : out;
    if (app < 3) {
      node_relu_kernel<<<(nfeat + 255) / 256, 256, 0, stream>>>(
          prev, agg, out, h16, nfeat);
    } else {
      node_softmax_kernel<<<(N_NODES * 32 + 255) / 256, 256, 0, stream>>>(
          prev, agg, out, N_NODES);
    }
  }
}
